// ImprovedTripleGraphModel_8246337209015
// MI455X (gfx1250) — matrix operand order repaired and checked by reading, not compiled, not run on hardware
//
#include <hip/hip_runtime.h>

#define N_NODES 50000
#define N_EDGES 400000
#define F0 128
#define F1 512
#define F2 1024
#define F3 2

#define BM 64
#define BN 128
#define BK 32

typedef __attribute__((ext_vector_type(16))) __bf16 v16bf;
typedef __attribute__((ext_vector_type(8)))  __bf16 v8bf;
typedef __attribute__((ext_vector_type(8)))  float  v8f;

__device__ __forceinline__ __bf16 bf_zero() {
  union { unsigned short s; __bf16 b; } o; o.s = 0; return o.b;
}

// ---------------------------------------------------------------- utilities
__global__ void fill_zero(float* __restrict__ p, int n) {
  int i = blockIdx.x * blockDim.x + threadIdx.x;
  if (i < n) p[i] = 0.0f;
}

__global__ void degree_kernel(const int* __restrict__ dst, float* __restrict__ deg, int e) {
  int i = blockIdx.x * blockDim.x + threadIdx.x;
  if (i < e) atomicAdd(&deg[dst[i]], 1.0f);
}

__global__ void rinv_kernel(float* __restrict__ d, int n) {
  int i = blockIdx.x * blockDim.x + threadIdx.x;
  if (i < n) d[i] = 1.0f / fmaxf(d[i], 1.0f);
}

// scatter-add h[src] into msg[dst]; one thread handles 4 consecutive features
__global__ void scatter_kernel(const float* __restrict__ h, const int* __restrict__ src,
                               const int* __restrict__ dst, float* __restrict__ msg,
                               int nedges, int F) {
  int idx = blockIdx.x * blockDim.x + threadIdx.x;
  int chunks = F >> 2;
  if (idx >= nedges * chunks) return;
  int e = idx / chunks;
  int f = (idx - e * chunks) << 2;
  int s = src[e], d = dst[e];
  float4 v = *(const float4*)(h + (size_t)s * F + f);
  float* mp = msg + (size_t)d * F + f;
  atomicAdd(mp + 0, v.x);
  atomicAdd(mp + 1, v.y);
  atomicAdd(mp + 2, v.z);
  atomicAdd(mp + 3, v.w);
}

// build bf16 concat weight Wbf[k][n], k in [0,2F): k<F -> W_l, else W_r
__global__ void convw_kernel(const float* __restrict__ Wl, const float* __restrict__ Wr,
                             __bf16* __restrict__ Wbf, int F, int Fout) {
  int i = blockIdx.x * blockDim.x + threadIdx.x;
  int total = 2 * F * Fout;
  if (i >= total) return;
  int k = i / Fout, n = i - k * Fout;
  float v = (k < F) ? Wl[(size_t)k * Fout + n] : Wr[(size_t)(k - F) * Fout + n];
  Wbf[i] = (__bf16)v;   // native v_cvt bf16 path
}

// ---------------------------------------------------------------- WMMA GEMM
// out[i,n] = relu( sum_k Acat[i,k]*Wbf[k,n] + bias[n] ),
// Acat[i, 0:F]   = msg[i,:] * rdeg[i]   (mean aggregation)
// Acat[i, F:2F]  = hin[i,:]
__global__ __launch_bounds__(256) void sage_gemm(
    const float* __restrict__ msg, const float* __restrict__ hin,
    const float* __restrict__ rdeg, const __bf16* __restrict__ Wbf,
    const float* __restrict__ bias, float* __restrict__ out,
    int nrows, int F, int Fout) {
  __shared__ __bf16 As[BM][BK];   // row-major [m][k]   4 KB
  __shared__ __bf16 Bs[BN][BK];   // transposed [n][k]  8 KB

  const int tid  = threadIdx.x;
  const int wave = tid >> 5;
  const int lane = tid & 31;
  const int half = lane >> 4;     // 0/1 : which 16-lane group
  const int l16  = lane & 15;
  const int m0 = blockIdx.x * BM;
  const int n0 = blockIdx.y * BN;
  const int K  = 2 * F;

  const int mi  = wave & 3;             // 0..3 : 16-row stripe
  const int niB = (wave >> 2) << 2;     // 0 or 4 : first of four 16-col tiles

  v8f acc[4] = {};

  // A staging: 64 rows x 32 k, 8 consecutive k per thread
  const int arow = tid >> 2;
  const int akk  = (tid & 3) << 3;
  // B staging: 128 n x 32 k, 16 consecutive k per thread
  const int brow = tid >> 1;
  const int bkk  = (tid & 1) << 4;

  const int gr = m0 + arow;
  const int gn = n0 + brow;
  const float rd = (gr < nrows) ? rdeg[gr] : 0.0f;

  for (int k0 = 0; k0 < K; k0 += BK) {
    // ---- stage A tile (fp32 -> bf16). BK=32 chunks never straddle the F boundary.
    {
      float va[8];
      int gk = k0 + akk;
      if (gr < nrows) {
        const float* sp;
        float sc;
        if (gk < F) { sp = msg + (size_t)gr * F + gk;        sc = rd;   }
        else        { sp = hin + (size_t)gr * F + (gk - F);  sc = 1.0f; }
        float4 p0 = ((const float4*)sp)[0];
        float4 p1 = ((const float4*)sp)[1];
        va[0] = p0.x * sc; va[1] = p0.y * sc; va[2] = p0.z * sc; va[3] = p0.w * sc;
        va[4] = p1.x * sc; va[5] = p1.y * sc; va[6] = p1.z * sc; va[7] = p1.w * sc;
      } else {
#pragma unroll
        for (int j = 0; j < 8; ++j) va[j] = 0.0f;
      }
      v8bf av;
#pragma unroll
      for (int j = 0; j < 8; ++j) av[j] = (__bf16)va[j];   // native packed cvt
      *(v8bf*)(&As[arow][akk]) = av;                       // one ds_store_b128
    }
    // ---- stage B tile transposed: Bs[n][k] <- Wbf[k][n]
    {
      v8bf b0, b1;
#pragma unroll
      for (int j = 0; j < 8; ++j) {
        int kk = k0 + bkk + j;
        b0[j] = (gn < Fout) ? Wbf[(size_t)kk * Fout + gn] : bf_zero();
      }
#pragma unroll
      for (int j = 0; j < 8; ++j) {
        int kk = k0 + bkk + 8 + j;
        b1[j] = (gn < Fout) ? Wbf[(size_t)kk * Fout + gn] : bf_zero();
      }
      *(v8bf*)(&Bs[brow][bkk])     = b0;   // 16B ds stores
      *(v8bf*)(&Bs[brow][bkk + 8]) = b1;
    }
    __syncthreads();

    // ---- A fragment (16x32 bf16): lanes 0-15 K=0..7 & 16..23, lanes 16-31 K=8..15 & 24..31
    const __bf16* ap = &As[mi * 16 + l16][half * 8];
    v8bf alo = *(const v8bf*)ap;
    v8bf ahi = *(const v8bf*)(ap + 16);
    v16bf afrag = __builtin_shufflevector(alo, ahi,
        0, 1, 2, 3, 4, 5, 6, 7, 8, 9, 10, 11, 12, 13, 14, 15);

    // ---- 4 B fragments + 4 WMMAs (B: lanes 0-15 K=0..7 & 16..23, lanes 16-31 K=8..15 & 24..31)
#pragma unroll
    for (int t = 0; t < 4; ++t) {
      const __bf16* bp = &Bs[(niB + t) * 16 + l16][half * 8];
      v8bf b0 = *(const v8bf*)bp;
      v8bf b1 = *(const v8bf*)(bp + 16);
      v16bf bfrag = __builtin_shufflevector(b0, b1,
          0, 1, 2, 3, 4, 5, 6, 7, 8, 9, 10, 11, 12, 13, 14, 15);
      acc[t] = __builtin_amdgcn_wmma_f32_16x16x32_bf16(
          false, afrag, false, bfrag, (short)0, acc[t], false, false);
    }
    __syncthreads();
  }

  // ---- epilogue: bias + relu + guarded store.  C layout: VGPR v -> M = half*8+v, lane&15 -> N
#pragma unroll
  for (int t = 0; t < 4; ++t) {
    int col = n0 + (niB + t) * 16 + l16;
    if (col >= Fout) continue;
    float bv = bias[col];
    v8f a = acc[t];
#pragma unroll
    for (int v = 0; v < 8; ++v) {
      int row = m0 + mi * 16 + half * 8 + v;
      if (row < nrows) out[(size_t)row * Fout + col] = fmaxf(a[v] + bv, 0.0f);
    }
  }
}

// ---------------------------------------------------------------- launcher
extern "C" void kernel_launch(void* const* d_in, const int* in_sizes, int n_in,
                              void* d_out, int out_size, void* d_ws, size_t ws_size,
                              hipStream_t stream) {
  (void)in_sizes; (void)n_in; (void)out_size; (void)ws_size;
  const float* x   = (const float*)d_in[0];
  const int*   ei  = (const int*)d_in[1];
  const float* Wl1 = (const float*)d_in[3];
  const float* bl1 = (const float*)d_in[4];
  const float* Wr1 = (const float*)d_in[5];
  const float* Wl2 = (const float*)d_in[6];
  const float* bl2 = (const float*)d_in[7];
  const float* Wr2 = (const float*)d_in[8];
  const float* Wl3 = (const float*)d_in[9];
  const float* bl3 = (const float*)d_in[10];
  const float* Wr3 = (const float*)d_in[11];
  float* out = (float*)d_out;

  const int* src = ei;            // edge_index row 0
  const int* dst = ei + N_EDGES;  // edge_index row 1

  // workspace carve (256B aligned)
  char* ws = (char*)d_ws;
  size_t off = 0;
  auto carve = [&](size_t bytes) -> void* {
    void* p = ws + off;
    off = (off + bytes + 255) & ~(size_t)255;
    return p;
  };
  float*  rdeg = (float*)carve((size_t)N_NODES * 4);
  float*  h1   = (float*)carve((size_t)N_NODES * F1 * 4);
  float*  h2   = (float*)carve((size_t)N_NODES * F2 * 4);
  float*  msg  = (float*)carve((size_t)N_NODES * F2 * 4);
  __bf16* Wbf  = (__bf16*)carve((size_t)2 * F2 * F2 * 2);

  const int gm = (N_NODES + BM - 1) / BM;  // 782
  dim3 blk(256);

  // degrees -> reciprocal
  fill_zero<<<(N_NODES + 255) / 256, 256, 0, stream>>>(rdeg, N_NODES);
  degree_kernel<<<(N_EDGES + 255) / 256, 256, 0, stream>>>(dst, rdeg, N_EDGES);
  rinv_kernel<<<(N_NODES + 255) / 256, 256, 0, stream>>>(rdeg, N_NODES);

  // ---- layer 1: 128 -> 512
  {
    const int F = F0, Fo = F1;
    int nm = N_NODES * F;
    fill_zero<<<(nm + 255) / 256, 256, 0, stream>>>(msg, nm);
    int tot = N_EDGES * (F >> 2);
    scatter_kernel<<<(tot + 255) / 256, 256, 0, stream>>>(x, src, dst, msg, N_EDGES, F);
    int wtot = 2 * F * Fo;
    convw_kernel<<<(wtot + 255) / 256, 256, 0, stream>>>(Wl1, Wr1, Wbf, F, Fo);
    sage_gemm<<<dim3(gm, (Fo + BN - 1) / BN), blk, 0, stream>>>(
        msg, x, rdeg, Wbf, bl1, h1, N_NODES, F, Fo);
  }
  // ---- layer 2: 512 -> 1024
  {
    const int F = F1, Fo = F2;
    int nm = N_NODES * F;
    fill_zero<<<(nm + 255) / 256, 256, 0, stream>>>(msg, nm);
    int tot = N_EDGES * (F >> 2);
    scatter_kernel<<<(tot + 255) / 256, 256, 0, stream>>>(h1, src, dst, msg, N_EDGES, F);
    int wtot = 2 * F * Fo;
    convw_kernel<<<(wtot + 255) / 256, 256, 0, stream>>>(Wl2, Wr2, Wbf, F, Fo);
    sage_gemm<<<dim3(gm, (Fo + BN - 1) / BN), blk, 0, stream>>>(
        msg, h1, rdeg, Wbf, bl2, h2, N_NODES, F, Fo);
  }
  // ---- layer 3: 1024 -> 2
  {
    const int F = F2, Fo = F3;
    int nm = N_NODES * F;
    fill_zero<<<(nm + 255) / 256, 256, 0, stream>>>(msg, nm);
    int tot = N_EDGES * (F >> 2);
    scatter_kernel<<<(tot + 255) / 256, 256, 0, stream>>>(h2, src, dst, msg, N_EDGES, F);
    int wtot = 2 * F * Fo;
    convw_kernel<<<(wtot + 255) / 256, 256, 0, stream>>>(Wl3, Wr3, Wbf, F, Fo);
    sage_gemm<<<dim3(gm, (Fo + BN - 1) / BN), blk, 0, stream>>>(
        msg, h2, rdeg, Wbf, bl3, out, N_NODES, F, Fo);
  }
}
